// _keypoint_decoder_56160992362716
// MI455X (gfx1250) — matrix operand order repaired and checked by reading, not compiled, not run on hardware
//
#include <hip/hip_runtime.h>

typedef __attribute__((ext_vector_type(16))) _Float16 v16h;
typedef __attribute__((ext_vector_type(8)))  float    v8f;

#define H_ 320
#define W_ 320

// Branch-free zero-padded fetch: clamped address + value select (clause-friendly)
__device__ __forceinline__ float feat_pad(const float* __restrict__ f, int b, int c, int y, int x) {
  bool valid = ((unsigned)y < (unsigned)H_) & ((unsigned)x < (unsigned)W_);
  int yc = min(max(y, 0), H_ - 1);
  int xc = min(max(x, 0), W_ - 1);
  float v = f[(((b * 2) + c) * H_ + yc) * W_ + xc];
  return valid ? v : 0.0f;
}

__global__ __launch_bounds__(256) void keypoint_fused(
    const float* __restrict__ features,  // (8,2,320,320)
    const float* __restrict__ off_w,     // (50,2,3,3)
    const float* __restrict__ off_b,     // (50)
    const float* __restrict__ dc_w,      // (1,2,5,5)
    const float* __restrict__ dc_b,      // (1)
    float* __restrict__ out)             // (8,1,320,320)
{
  // [wave][pixel][offset channel]; stride 64 so the D-scatter needs no guard
  __shared__ float soff[8][16][64];                       // 32 KB
  __shared__ __align__(16) _Float16 aLds[4][32][16];      // pre-swizzled A frags, 4 KB
  __shared__ float sbias[64];                             // off_b padded with zeros
  __shared__ float sdcw[64];                              // dc_w (50) padded

  const int tid  = threadIdx.x;
  const int wave = tid >> 5;
  const int lane = tid & 31;
  const int col  = lane & 15;
  const bool hi  = lane >= 16;

  // ---- Cooperative one-time staging (uniform across the block) ----
  // A-fragment k order chosen here (contiguous, same as bfrag): lane<16 row M=lane: e0..15 -> K=0..15 ;
  // lane>=16 row M=lane-16: e0..15 -> K=16..31.
#pragma unroll
  for (int i = tid; i < 4 * 32 * 16; i += 256) {
    int m  = i >> 9;
    int ln = (i >> 4) & 31;
    int e  = i & 15;
    int oc = 16 * m + (ln & 15);
    int K  = e + ((ln >= 16) ? 16 : 0);
    bool valid = (oc < 50) && (K < 18);
    float v = off_w[valid ? (oc * 18 + K) : 0];
    aLds[m][ln][e] = valid ? (_Float16)v : (_Float16)0.0f;
  }
  for (int i = tid; i < 64; i += 256) {
    sbias[i] = (i < 50) ? off_b[i] : 0.0f;
    sdcw[i]  = (i < 50) ? dc_w[i]  : 0.0f;
  }

  // wave-tile -> (batch, row, 16-pixel segment); 320/16 = 20 tiles per row
  const int tile = blockIdx.x * 8 + wave;
  const int b    = tile / 6400;          // 320 rows * 20 tiles
  const int r0   = tile % 6400;
  const int y    = r0 / 20;
  const int x0   = (r0 % 20) * 16;
  const int px   = x0 + col;

  __builtin_prefetch(features + ((b * 2) * H_ + y) * W_ + px, 0, 1);

  // ---- Stage 1: im2col B fragment (32x16 f16). K = c*9 + dy*3 + dx (18 real) ----
  // Lanes 0-15 carry K=0..15 for pixel `col`; lanes 16-31 carry K=16..31 (16,17 real).
  v16h bfrag;
#pragma unroll
  for (int e = 0; e < 16; ++e) bfrag[e] = (_Float16)0.0f;
  if (!hi) {
#pragma unroll
    for (int e = 0; e < 16; ++e) {              // K = e, constants fold after unroll
      int c = e / 9, t = e % 9;
      bfrag[e] = (_Float16)feat_pad(features, b, c, y + t / 3 - 1, px + t % 3 - 1);
    }
  } else {
    bfrag[0] = (_Float16)feat_pad(features, b, 1, y + 1, px);      // K=16: c1,dy2,dx1
    bfrag[1] = (_Float16)feat_pad(features, b, 1, y + 1, px + 1);  // K=17: c1,dy2,dx2
  }

  __syncthreads();  // aLds/sbias/sdcw ready

  // ---- 4 WMMAs: offset channels 0..63 (50 real) for 16 pixels ----
  v8f acc[4];
#pragma unroll
  for (int m = 0; m < 4; ++m) {
    v16h afrag = *reinterpret_cast<const v16h*>(&aLds[m][lane][0]);  // 2x ds_load_b128
    v8f c0 = {};
    acc[m] = __builtin_amdgcn_wmma_f32_16x16x32_f16(false, afrag, false, bfrag,
                                                    (short)0, c0, false, false);
  }

  // D layout: VGPR r, lane<16 -> (M=r,N=lane); lane>=16 -> (M=r+8,N=lane-16).
  // Unguarded scatter (channels 50..63 are junk lanes of the padded stride).
#pragma unroll
  for (int m = 0; m < 4; ++m) {
#pragma unroll
    for (int r = 0; r < 8; ++r) {
      int oc = 16 * m + (hi ? 8 : 0) + r;
      soff[wave][col][oc] = acc[m][r] + sbias[oc];
    }
  }
  __syncthreads();

  // ---- Stage 2: 5x5 deformable conv (C=2 -> 1); two lanes split the 25 taps ----
  float part = 0.0f;
  const int klo = hi ? 13 : 0;
  const int khi = hi ? 25 : 13;
  for (int k = klo; k < khi; ++k) {
    float oy = soff[wave][col][2 * k];
    float ox = soff[wave][col][2 * k + 1];
    int ky = k / 5, kx = k % 5;
    float py  = (float)(y + ky - 2) + oy;
    float pxf = (float)(px + kx - 2) + ox;
    float y0f = floorf(py), x0f = floorf(pxf);
    float ly = py - y0f, lx = pxf - x0f;
    int y0 = (int)y0f, xi0 = (int)x0f;
    float w00 = (1.0f - ly) * (1.0f - lx), w01 = (1.0f - ly) * lx;
    float w10 = ly * (1.0f - lx),          w11 = ly * lx;
#pragma unroll
    for (int c = 0; c < 2; ++c) {
      float v00 = feat_pad(features, b, c, y0,     xi0);
      float v01 = feat_pad(features, b, c, y0,     xi0 + 1);
      float v10 = feat_pad(features, b, c, y0 + 1, xi0);
      float v11 = feat_pad(features, b, c, y0 + 1, xi0 + 1);
      float bil = v00 * w00 + v01 * w01 + v10 * w10 + v11 * w11;
      part = fmaf(sdcw[c * 25 + k], bil, part);
    }
  }
  float total = part + __shfl_xor(part, 16, 32);  // wave32 cross-half reduce
  float score = total + dc_b[0];
  asm volatile("" :: "v"(score));  // keep the WMMA + gather pipeline live

  // BN (finite affine) + softmax over the singleton channel axis == 1.0 exactly
  // for finite scores; fma(s,0,1) reproduces that, NaN/Inf-propagating.
  if (!hi)
    out[(b * H_ + y) * W_ + px] = __builtin_fmaf(score, 0.0f, 1.0f);
}

extern "C" void kernel_launch(void* const* d_in, const int* in_sizes, int n_in,
                              void* d_out, int out_size, void* d_ws, size_t ws_size,
                              hipStream_t stream) {
  (void)in_sizes; (void)n_in; (void)out_size; (void)d_ws; (void)ws_size;
  const float* features = (const float*)d_in[0];
  const float* off_w    = (const float*)d_in[1];
  const float* off_b    = (const float*)d_in[2];
  const float* dc_w     = (const float*)d_in[3];
  const float* dc_b     = (const float*)d_in[4];
  float* out = (float*)d_out;

  // 8 batches * 320 rows * 20 tiles-of-16 = 51200 wave-tiles; 8 waves/block.
  dim3 grid(6400), block(256);
  hipLaunchKernelGGL(keypoint_fused, grid, block, 0, stream,
                     features, off_w, off_b, dc_w, dc_b, out);
}
